// UMPNodeBlock_38809324487019
// MI455X (gfx1250) — matrix operand order repaired and checked by reading, not compiled, not run on hardware
//
#include <hip/hip_runtime.h>
#include <hip/hip_bf16.h>
#include <math.h>

typedef __attribute__((ext_vector_type(16))) __bf16 v16bf;
typedef __attribute__((ext_vector_type(8)))  float  v8f;

#define DH 128
#define STRIPES 64

// ---------------------------------------------------------------------------
// WMMA K-loop: A tile (16 x K bf16) in LDS, B column (K bf16, transposed
// weights) in global.  K order used here (contiguous, same for A and B):
//   A 16x32 bf16: lane l -> row m=l&15; l<16 holds K=[kk..kk+15]
//                 l>=16 holds K=[kk+16..kk+31]
//   B 32x16 bf16: lane l -> col n=l&15; l<16 holds K=[kk..kk+15], l>=16 K=[kk+16..kk+31]
// ---------------------------------------------------------------------------
template<int K, int STRIDE>
__device__ __forceinline__ v8f wmma_ktile(const __bf16* As, const __bf16* Bcol, int lane) {
    const int m  = lane & 15;
    const int hi = lane >> 4;
    v8f acc = (v8f){0.f, 0.f, 0.f, 0.f, 0.f, 0.f, 0.f, 0.f};
    __builtin_prefetch(Bcol, 0, 1);  // global_prefetch for the weight column
#pragma unroll
    for (int kk = 0; kk < K; kk += 32) {
        v16bf a, b;
        const __bf16* ap = As + m * STRIDE + kk + 16 * hi;
#pragma unroll
        for (int j = 0; j < 8; ++j) a[j]     = ap[j];
#pragma unroll
        for (int j = 0; j < 8; ++j) a[8 + j] = ap[8 + j];
        const __bf16* bp = Bcol + kk + 16 * hi;
#pragma unroll
        for (int j = 0; j < 16; ++j) b[j] = bp[j];
        acc = __builtin_amdgcn_wmma_f32_16x16x32_bf16(false, a, false, b,
                                                      (short)0, acc, false, false);
    }
    return acc;
}

// Epilogue: add bias, store pre-BN activation, accumulate striped column
// sum / sum-of-squares for the training-mode BatchNorm that follows.
template<typename OT>
__device__ __forceinline__ void epilogue_bias_stats(v8f acc, float bias, int tile, int n,
                                                    int lane, int rows,
                                                    OT* __restrict__ outp,
                                                    float* __restrict__ stats, int stripe) {
    const int hi = lane >> 4;
    float s1 = 0.f, s2 = 0.f;
#pragma unroll
    for (int r = 0; r < 8; ++r) {
        const int row = tile * 16 + r + 8 * hi;
        if (row < rows) {
            float v = acc[r] + bias;
            outp[(size_t)row * DH + n] = (OT)v;
            s1 += v;
            s2 += v * v;
        }
    }
    s1 += __shfl_xor(s1, 16, 32);   // lanes l and l^16 share column n
    s2 += __shfl_xor(s2, 16, 32);
    if (lane < 16) {
        float* sp = stats + (size_t)stripe * 256;
        atomicAdd(sp + n, s1);
        atomicAdd(sp + 128 + n, s2);
    }
}

// ---------------------------------------------------------------------------
// Weight prep: fp32 [K,128] -> bf16 transposed [128,K] so each WMMA lane
// reads its output column contiguously.
// segments: w1aT @0 (K=256), w1bT @32768 (K=128), w2aT @49152 (K=256), w2bT @81920 (K=128)
// ---------------------------------------------------------------------------
__global__ void __launch_bounds__(256)
prep_weights(const float* __restrict__ w1a, const float* __restrict__ w1b,
             const float* __restrict__ w2a, const float* __restrict__ w2b,
             __bf16* __restrict__ wT) {
    int idx = blockIdx.x * 256 + threadIdx.x;
    const float* src; int K, base;
    if      (idx < 32768) { src = w1a; K = 256; base = 0;     }
    else if (idx < 49152) { src = w1b; K = 128; base = 32768; }
    else if (idx < 81920) { src = w2a; K = 256; base = 49152; }
    else if (idx < 98304) { src = w2b; K = 128; base = 81920; }
    else return;
    int li = idx - base;
    int n = li / K, k = li % K;
    wT[idx] = (__bf16)src[(size_t)k * 128 + n];
}

// ---------------------------------------------------------------------------
// GEMM 1 (edges, K=256): A = concat(message[e], x[row[e]]) fused into staging.
// ---------------------------------------------------------------------------
__global__ void __launch_bounds__(256)
gemm_edge_a(const float* __restrict__ msg, const float* __restrict__ x,
            const int* __restrict__ erow, const __bf16* __restrict__ wT,
            const float* __restrict__ bias, __bf16* __restrict__ h1,
            float* __restrict__ stats, int E) {
    __shared__ __bf16 As[16 * 264];
    const int t = threadIdx.x, tile = blockIdx.x;
    {
        const int r = t >> 4, cs = (t & 15) * 16;
        const int e = tile * 16 + r;
        __bf16* dst = As + r * 264 + cs;
        if (e < E) {
            const float* srcp = (cs < 128)
                ? (msg + (size_t)e * 128 + cs)
                : (x + (size_t)erow[e] * 128 + (cs - 128));
#pragma unroll
            for (int j = 0; j < 16; ++j) dst[j] = (__bf16)srcp[j];
        } else {
#pragma unroll
            for (int j = 0; j < 16; ++j) dst[j] = (__bf16)0.f;
        }
    }
    __syncthreads();
    const int wv = t >> 5, lane = t & 31;
    const int n = wv * 16 + (lane & 15);
    v8f acc = wmma_ktile<256, 264>(As, wT + (size_t)n * 256, lane);
    epilogue_bias_stats<__bf16>(acc, bias[n], tile, n, lane, E, h1, stats,
                                blockIdx.x & (STRIPES - 1));
}

// ---------------------------------------------------------------------------
// GEMM for hidden layers (K=128): A = relu(bn(h_in)) fused into staging.
// Used for edge layer 1b (OT=__bf16) and node layer 2b (OT=float).
// ---------------------------------------------------------------------------
template<typename OT>
__global__ void __launch_bounds__(256)
gemm_bn_hid(const __bf16* __restrict__ hin, const float* __restrict__ bn,
            const __bf16* __restrict__ wT, const float* __restrict__ bias,
            OT* __restrict__ hout, float* __restrict__ stats, int rows) {
    __shared__ __bf16 As[16 * 136];
    const int t = threadIdx.x, tile = blockIdx.x;
    {
        const int r = t >> 4, c0 = (t & 15) * 8;
        const int e = tile * 16 + r;
        __bf16* dst = As + r * 136 + c0;
        if (e < rows) {
#pragma unroll
            for (int j = 0; j < 8; ++j) {
                int c = c0 + j;
                float v = fmaxf((float)hin[(size_t)e * DH + c] * bn[c] + bn[128 + c], 0.f);
                dst[j] = (__bf16)v;
            }
        } else {
#pragma unroll
            for (int j = 0; j < 8; ++j) dst[j] = (__bf16)0.f;
        }
    }
    __syncthreads();
    const int wv = t >> 5, lane = t & 31;
    const int n = wv * 16 + (lane & 15);
    v8f acc = wmma_ktile<128, 136>(As, wT + (size_t)n * 128, lane);
    epilogue_bias_stats<OT>(acc, bias[n], tile, n, lane, rows, hout, stats,
                            blockIdx.x & (STRIPES - 1));
}

// ---------------------------------------------------------------------------
// GEMM 3 (nodes, K=256): A = concat(x[node], s[node]/max(cnt,1)) fused.
// ---------------------------------------------------------------------------
__global__ void __launch_bounds__(256)
gemm_node_a(const float* __restrict__ x, const float* __restrict__ s,
            const float* __restrict__ cnt, const __bf16* __restrict__ wT,
            const float* __restrict__ bias, __bf16* __restrict__ o1,
            float* __restrict__ stats, int N) {
    __shared__ __bf16 As[16 * 264];
    const int t = threadIdx.x, tile = blockIdx.x;
    {
        const int r = t >> 4, cs = (t & 15) * 16;
        const int node = tile * 16 + r;
        __bf16* dst = As + r * 264 + cs;
        if (node < N) {
            if (cs < 128) {
                const float* sp = x + (size_t)node * 128 + cs;
#pragma unroll
                for (int j = 0; j < 16; ++j) dst[j] = (__bf16)sp[j];
            } else {
                float inv = 1.f / fmaxf(cnt[node], 1.f);
                const float* sp = s + (size_t)node * 128 + (cs - 128);
#pragma unroll
                for (int j = 0; j < 16; ++j) dst[j] = (__bf16)(sp[j] * inv);
            }
        } else {
#pragma unroll
            for (int j = 0; j < 16; ++j) dst[j] = (__bf16)0.f;
        }
    }
    __syncthreads();
    const int wv = t >> 5, lane = t & 31;
    const int n = wv * 16 + (lane & 15);
    v8f acc = wmma_ktile<256, 264>(As, wT + (size_t)n * 256, lane);
    epilogue_bias_stats<__bf16>(acc, bias[n], tile, n, lane, N, o1, stats,
                                blockIdx.x & (STRIPES - 1));
}

// ---------------------------------------------------------------------------
// Reduce striped column stats -> per-column BN scale/shift.
// ---------------------------------------------------------------------------
__global__ void __launch_bounds__(128)
bn_stats(const float* __restrict__ stats, const float* __restrict__ gamma,
         const float* __restrict__ beta, float* __restrict__ bn, float count) {
    int c = threadIdx.x;
    float su = 0.f, sq = 0.f;
    for (int s = 0; s < STRIPES; ++s) {
        su += stats[(size_t)s * 256 + c];
        sq += stats[(size_t)s * 256 + 128 + c];
    }
    float mean  = su / count;
    float var   = sq / count - mean * mean;
    float scale = gamma[c] * rsqrtf(var + 1e-5f);
    bn[c]       = scale;
    bn[128 + c] = beta[c] - mean * scale;
}

// ---------------------------------------------------------------------------
// Scatter-mean numerator + count: s[col[e]] += relu(bn(h2[e])), cnt[col[e]] += 1
// ---------------------------------------------------------------------------
__global__ void __launch_bounds__(256)
scatter_edges(const __bf16* __restrict__ h2, const int* __restrict__ ecol,
              const float* __restrict__ bn, float* __restrict__ s,
              float* __restrict__ cnt, int E) {
    int tid = blockIdx.x * 256 + threadIdx.x;
    int e = tid >> 7, c = tid & 127;
    if (e >= E) return;
    int dst = ecol[e];
    float v = fmaxf((float)h2[(size_t)e * DH + c] * bn[c] + bn[128 + c], 0.f);
    atomicAdd(s + (size_t)dst * DH + c, v);
    if (c == 0) atomicAdd(cnt + dst, 1.0f);
}

// ---------------------------------------------------------------------------
// Final: o = relu(bn(o2pre)) -> out[0:N*128]; attn = sigmoid(o . wa + ba)
// ---------------------------------------------------------------------------
__global__ void __launch_bounds__(128)
final_out(const float* __restrict__ o2, const float* __restrict__ bn,
          const float* __restrict__ wa, const float* __restrict__ ba,
          float* __restrict__ out, int N) {
    __shared__ float red[4];
    int i = blockIdx.x, c = threadIdx.x;
    float v = fmaxf(o2[(size_t)i * 128 + c] * bn[c] + bn[128 + c], 0.f);
    out[(size_t)i * 128 + c] = v;
    float p = v * wa[c];
#pragma unroll
    for (int d = 16; d > 0; d >>= 1) p += __shfl_xor(p, d, 32);
    if ((c & 31) == 0) red[c >> 5] = p;
    __syncthreads();
    if (c == 0) {
        float dsum = red[0] + red[1] + red[2] + red[3] + ba[0];
        out[(size_t)N * 128 + i] = 1.f / (1.f + __expf(-dsum));
    }
}

// ---------------------------------------------------------------------------
extern "C" void kernel_launch(void* const* d_in, const int* in_sizes, int n_in,
                              void* d_out, int out_size, void* d_ws, size_t ws_size,
                              hipStream_t stream) {
    const float* x    = (const float*)d_in[0];
    const int*   ei   = (const int*)d_in[1];
    const float* msg  = (const float*)d_in[2];
    const float* w1a  = (const float*)d_in[3];
    const float* b1a  = (const float*)d_in[4];
    const float* g1a  = (const float*)d_in[5];
    const float* be1a = (const float*)d_in[6];
    const float* w1b  = (const float*)d_in[7];
    const float* b1b  = (const float*)d_in[8];
    const float* g1b  = (const float*)d_in[9];
    const float* be1b = (const float*)d_in[10];
    const float* w2a  = (const float*)d_in[11];
    const float* b2a  = (const float*)d_in[12];
    const float* g2a  = (const float*)d_in[13];
    const float* be2a = (const float*)d_in[14];
    const float* w2b  = (const float*)d_in[15];
    const float* b2b  = (const float*)d_in[16];
    const float* g2b  = (const float*)d_in[17];
    const float* be2b = (const float*)d_in[18];
    const float* wa   = (const float*)d_in[19];
    const float* ba   = (const float*)d_in[20];
    float* out = (float*)d_out;

    const int N = in_sizes[0] / 128;   // 50000
    const int E = in_sizes[1] / 2;     // 800000
    const int* erow = ei;
    const int* ecol = ei + E;

    // ---- workspace carve-out (256B aligned) ----
    char* ws = (char*)d_ws;
    size_t off = 0;
    auto carve = [&](size_t bytes) {
        size_t cur = off;
        off = (off + bytes + 255) & ~(size_t)255;
        return cur;
    };
    size_t off_stats = carve((size_t)4 * STRIPES * 256 * 4);  // 4 BN layers
    size_t off_bn    = carve((size_t)4 * 256 * 4);            // scale/shift x4
    size_t off_wT    = carve((size_t)98304 * 2);              // bf16 transposed weights
    size_t off_s     = carve((size_t)N * DH * 4);             // scatter numerator
    size_t off_cnt   = carve((size_t)N * 4);                  // scatter counts
    size_t off_h1    = carve((size_t)E * DH * 2);             // edge pre-BN act 1 (bf16)
    size_t off_h2    = carve((size_t)E * DH * 2);             // edge pre-BN act 2 (bf16)
    size_t off_o1    = carve((size_t)N * DH * 2);             // node pre-BN act 1 (bf16)
    size_t off_o2    = carve((size_t)N * DH * 4);             // node pre-BN act 2 (f32)
    (void)ws_size;

    float*  stats1 = (float*)(ws + off_stats);
    float*  stats2 = stats1 + (size_t)STRIPES * 256;
    float*  stats3 = stats2 + (size_t)STRIPES * 256;
    float*  stats4 = stats3 + (size_t)STRIPES * 256;
    float*  bn1 = (float*)(ws + off_bn);
    float*  bn2 = bn1 + 256;
    float*  bn3 = bn2 + 256;
    float*  bn4 = bn3 + 256;
    __bf16* wT   = (__bf16*)(ws + off_wT);
    __bf16* w1aT = wT;
    __bf16* w1bT = wT + 32768;
    __bf16* w2aT = wT + 49152;
    __bf16* w2bT = wT + 81920;
    float*  sbuf = (float*)(ws + off_s);
    float*  cnt  = (float*)(ws + off_cnt);
    __bf16* h1   = (__bf16*)(ws + off_h1);
    __bf16* h2   = (__bf16*)(ws + off_h2);
    __bf16* o1   = (__bf16*)(ws + off_o1);
    float*  o2   = (float*)(ws + off_o2);

    // zero accumulators every call (graph-replay safe)
    hipMemsetAsync(ws + off_stats, 0, (size_t)4 * STRIPES * 256 * 4, stream);
    hipMemsetAsync(ws + off_s,     0, (size_t)N * DH * 4, stream);
    hipMemsetAsync(ws + off_cnt,   0, (size_t)N * 4, stream);

    const int eTiles = (E + 15) / 16;   // 50000
    const int nTiles = (N + 15) / 16;   // 3125

    prep_weights<<<384, 256, 0, stream>>>(w1a, w1b, w2a, w2b, wT);

    // edge MLP layer 1: [E,256] @ [256,128]
    gemm_edge_a<<<eTiles, 256, 0, stream>>>(msg, x, erow, w1aT, b1a, h1, stats1, E);
    bn_stats<<<1, 128, 0, stream>>>(stats1, g1a, be1a, bn1, (float)E);

    // edge MLP layer 2: [E,128] @ [128,128]  (BN1+ReLU fused into A load)
    gemm_bn_hid<__bf16><<<eTiles, 256, 0, stream>>>(h1, bn1, w1bT, b1b, h2, stats2, E);
    bn_stats<<<1, 128, 0, stream>>>(stats2, g1b, be1b, bn2, (float)E);

    // scatter-mean (BN2+ReLU fused), numerator + counts
    scatter_edges<<<(int)(((size_t)E * DH + 255) / 256), 256, 0, stream>>>(
        h2, ecol, bn2, sbuf, cnt, E);

    // node MLP layer 1: [N,256] @ [256,128]  (concat + mean fused into A load)
    gemm_node_a<<<nTiles, 256, 0, stream>>>(x, sbuf, cnt, w2aT, b2a, o1, stats3, N);
    bn_stats<<<1, 128, 0, stream>>>(stats3, g2a, be2a, bn3, (float)N);

    // node MLP layer 2: [N,128] @ [128,128]  (BN3+ReLU fused into A load)
    gemm_bn_hid<float><<<nTiles, 256, 0, stream>>>(o1, bn3, w2bT, b2b, o2, stats4, N);
    bn_stats<<<1, 128, 0, stream>>>(stats4, g2b, be2b, bn4, (float)N);

    // final BN4+ReLU -> o, plus attention head
    final_out<<<N, 128, 0, stream>>>(o2, bn4, wa, ba, out, N);
}
